// FcGagaLayer_88433376625357
// MI455X (gfx1250) — matrix operand order repaired and checked by reading, not compiled, not run on hardware
//
#include <hip/hip_runtime.h>
#include <hip/hip_bf16.h>

typedef __bf16 bf16;
typedef __attribute__((ext_vector_type(16))) __bf16 v16bf;
typedef __attribute__((ext_vector_type(8)))  __bf16 v8bf;
typedef __attribute__((ext_vector_type(2)))  __bf16 v2bf;
typedef __attribute__((ext_vector_type(8)))  float  v8f;
typedef __attribute__((ext_vector_type(4)))  unsigned int u32x4;
typedef __attribute__((ext_vector_type(8)))  int i32x8;
typedef __attribute__((ext_vector_type(4)))  int i32x4;

#if __has_builtin(__builtin_amdgcn_tensor_load_to_lds) && \
    __has_builtin(__builtin_amdgcn_s_wait_tensorcnt)
#define USE_TDM 1
#else
#define USE_TDM 0
#endif

// Problem constants
constexpr int B_   = 64;
constexpr int N_   = 207;
constexpr int H_   = 12;
constexpr int HOR_ = 12;
constexpr int D_   = 64;
constexpr int HID_ = 128;
constexpr int IN_SIZE_ = H_ * (N_ + 1) + D_;   // 2560
constexpr int ROWS_ = B_ * N_;                  // 13248
constexpr float EPS_ = 10.0f;

// ---------------------------------------------------------------------------
// WMMA helpers (CDNA5 wave32 layouts, bf16 operands, K=32, f32 accum)
// ---------------------------------------------------------------------------
__device__ inline v16bf frag_concat(v8bf lo, v8bf hv) {
  return __builtin_shufflevector(lo, hv, 0, 1, 2, 3, 4, 5, 6, 7,
                                 8, 9, 10, 11, 12, 13, 14, 15);
}

__device__ inline v16bf load_a_frag(const bf16* rowp, int hi) {
  v8bf lo = *(const v8bf*)(rowp + hi * 16);
  v8bf hv = *(const v8bf*)(rowp + hi * 16 + 8);
  return frag_concat(lo, hv);
}

__device__ inline v16bf load_b_frag(const bf16* chunk) {
  v8bf lo = *(const v8bf*)(chunk);
  v8bf hv = *(const v8bf*)(chunk + 8);
  return frag_concat(lo, hv);
}

// Convert 16 strided f32 weights into one lane-chunk (two 16B LDS stores).
__device__ inline void fill_chunk16(bf16* dst, const float* src, int stride) {
  v8bf a, b;
#pragma unroll
  for (int p = 0; p < 8; ++p) a[p] = (bf16)src[p * stride];
#pragma unroll
  for (int p = 0; p < 8; ++p) b[p] = (bf16)src[(p + 8) * stride];
  *(v8bf*)dst = a;
  *(v8bf*)(dst + 8) = b;
}

__device__ inline v8f wmma_bf16(v16bf a, v16bf b, v8f c) {
  return __builtin_amdgcn_wmma_f32_16x16x32_bf16(false, a, false, b,
                                                 (short)0, c, false, false);
}

#if USE_TDM
// TDM: DMA a 32x32 bf16 tile (row stride 2560 elems) from global into LDS,
// LDS row stride = 40 bf16 (pad: 4 dwords after every 16 dwords).
// Descriptor per CDNA5 ISA.
// Toolchain uses the 6-arg builtin: (u32x4, i32x8, i32x4, i32x4, i32x8, i32).
__device__ inline void tdm_load_a_tile(unsigned int lds_addr, const bf16* gsrc) {
  unsigned long long ga = (unsigned long long)gsrc;
  u32x4 g0;
  g0[0] = 1u;                                   // count=1 (valid), user mode
  g0[1] = lds_addr;                             // lds_addr (bytes)
  g0[2] = (unsigned int)ga;                     // global_addr[31:0]
  g0[3] = (unsigned int)(ga >> 32) | (2u << 30);// global_addr[56:32] | type=2
  i32x8 g1;
  g1[0] = (1 << 16)      // data_size = 1 -> 2 bytes
        | (1 << 20)      // pad_enable
        | (3 << 22)      // pad_interval: code 3 -> every 16 dwords (64B row)
        | (3 << 25);     // pad_amount:   code 3 -> 4 dwords (16B) padding
  g1[1] = (int)((IN_SIZE_ & 0xffff) << 16);               // tensor_dim0 lo16
  g1[2] = (int)((IN_SIZE_ >> 16) | ((ROWS_ & 0xffff) << 16)); // dim0 hi, dim1 lo
  g1[3] = (int)((ROWS_ >> 16) | (32 << 16));              // dim1 hi, tile_dim0=32
  g1[4] = 32;                                             // tile_dim1=32, tile_dim2=0
  g1[5] = IN_SIZE_;                                       // tensor_dim0_stride lo32
  g1[6] = 0;                                              // stride hi, dim1_stride lo
  g1[7] = 0;
  i32x4 z4;
  z4[0] = 0; z4[1] = 0; z4[2] = 0; z4[3] = 0;
  i32x8 z8;
#pragma unroll
  for (int i = 0; i < 8; ++i) z8[i] = 0;
  __builtin_amdgcn_tensor_load_to_lds(g0, g1, z4, z4, z8, 0);
}
#endif

// ---------------------------------------------------------------------------
// Kernel 1: tg = relu(concat(emb[nid], tod) @ Wg1 + bg1)   (ROWS x HID)
// ---------------------------------------------------------------------------
__global__ __launch_bounds__(128) void gates1_kernel(
    const float* __restrict__ emb, const int* __restrict__ nid,
    const float* __restrict__ tod, const float* __restrict__ Wg1,
    const float* __restrict__ bg1, float* __restrict__ tg) {
  int row = blockIdx.x;
  int c = threadIdx.x;
  int e = nid[row];
  float s = bg1[c];
  const float* er = emb + e * D_;
#pragma unroll 4
  for (int k = 0; k < D_; ++k) s += er[k] * Wg1[k * HID_ + c];
  const float* tr = tod + row * H_;
#pragma unroll
  for (int k = 0; k < H_; ++k) s += tr[k] * Wg1[(D_ + k) * HID_ + c];
  tg[row * HID_ + c] = s > 0.f ? s : 0.f;
}

// ---------------------------------------------------------------------------
// Kernel 2: per row: tgf, tgb, hist = history/(1+tgb), level; zero facc
// ---------------------------------------------------------------------------
__global__ __launch_bounds__(128) void gates2_kernel(
    const float* __restrict__ tg, const float* __restrict__ Wg2,
    const float* __restrict__ bg2, const float* __restrict__ Wg3,
    const float* __restrict__ bg3, const float* __restrict__ history,
    float* __restrict__ tgf, float* __restrict__ tgb,
    float* __restrict__ hist, float* __restrict__ level,
    float* __restrict__ facc) {
  int row = blockIdx.x * blockDim.x + threadIdx.x;
  if (row >= ROWS_) return;
  const float* tr = tg + row * HID_;
  float lev = -3.4e38f;
  for (int c = 0; c < HOR_; ++c) {
    float sf = bg2[c];
    float sb = bg3[c];
    for (int k = 0; k < HID_; ++k) {
      float t = tr[k];
      sf += t * Wg2[k * HOR_ + c];
      sb += t * Wg3[k * H_ + c];
    }
    tgf[row * HOR_ + c] = sf;
    tgb[row * H_ + c] = sb;
    float hv = history[row * H_ + c] / (1.0f + sb);
    hist[row * H_ + c] = hv;
    lev = fmaxf(lev, hv);
    facc[row * HOR_ + c] = 0.0f;
  }
  level[row] = lev;
}

// ---------------------------------------------------------------------------
// Kernel 3: dp = exp(EPS * E E^T)   (N x N)
// ---------------------------------------------------------------------------
__global__ __launch_bounds__(256) void dp_kernel(
    const float* __restrict__ emb, const int* __restrict__ nid,
    float* __restrict__ dp) {
  int idx = blockIdx.x * blockDim.x + threadIdx.x;
  if (idx >= N_ * N_) return;
  int n = idx / N_, m = idx % N_;
  const float* a = emb + nid[n] * D_;
  const float* b = emb + nid[m] * D_;
  float s = 0.f;
#pragma unroll 4
  for (int d = 0; d < D_; ++d) s += a[d] * b[d];
  dp[idx] = expf(EPS_ * s);
}

// ---------------------------------------------------------------------------
// Kernel 4: assemble h into backcast buffer (= d_out) + optional bf16 mirror
// ---------------------------------------------------------------------------
__global__ __launch_bounds__(256) void hbuild_kernel(
    const float* __restrict__ hist, const float* __restrict__ level,
    const float* __restrict__ dp, const float* __restrict__ emb,
    const int* __restrict__ nid, float* __restrict__ bc,
    bf16* __restrict__ bch) {
  int row = blockIdx.x;
  int b = row / N_, n = row % N_;
  float lev = level[row];
  float safe = (lev == 0.f) ? 1.f : lev;
  float inv = 1.0f / safe;
  bool z = (lev == 0.f);
  float* orow = bc + (size_t)row * IN_SIZE_;
  bf16* brow = bch ? bch + (size_t)row * IN_SIZE_ : nullptr;
  for (int e = threadIdx.x; e < IN_SIZE_; e += 256) {
    float val;
    if (e < H_) {
      val = z ? 0.f : hist[row * H_ + e] * inv;
    } else if (e < H_ + N_ * H_) {
      int ee = e - H_;
      int m = ee / H_, hh = ee % H_;
      float v2 = hist[(b * N_ + m) * H_ + hh] * dp[n * N_ + m];
      val = z ? 0.f : fmaxf((v2 - lev) * inv, 0.f);
    } else {
      val = emb[nid[row] * D_ + (e - H_ - N_ * H_)];
    }
    orow[e] = val;
    if (brow) brow[e] = (bf16)val;
  }
}

// ---------------------------------------------------------------------------
// Kernel 5: one N-BEATS block, fused, WMMA bf16->f32, TDM-fed A tiles.
// Grid: 414 workgroups x 128 threads (4 waves). Each WG owns 32 rows.
// ---------------------------------------------------------------------------
__global__ __launch_bounds__(128) void block_kernel(
    const float* __restrict__ bc_in, float* __restrict__ bc_out,
    const bf16* __restrict__ bch_in, bf16* __restrict__ bch_out,
    const float* __restrict__ Wfc0, const float* __restrict__ bfc0,
    const float* __restrict__ Wfc,  const float* __restrict__ bfc,
    const float* __restrict__ Wf,   const float* __restrict__ bfv,
    const float* __restrict__ Wb,   const float* __restrict__ bb,
    float* __restrict__ facc) {
  __shared__ __align__(16) bf16 sA[32][40];              // A K-tile (row major)
  __shared__ __align__(16) bf16 sWfrag[4 * 8 * 32 * 16]; // weights, frag order
  __shared__ __align__(16) bf16 sX[2][32][136];          // x ping-pong

  const int tid = threadIdx.x;
  const int wave = tid >> 5;
  const int lane = tid & 31;
  const int l = lane & 15;
  const int hi = lane >> 4;
  const int rowTile = wave & 1;               // 16-row strip
  const int ct0 = (wave >> 1) * 4;            // first col tile (0 or 4)
  const int mBase = blockIdx.x * 32;
  const bool has_mirror = (bch_in != nullptr);

  // ---------------- Stage 1: x = relu(backcast @ Wfc0 + bfc0) --------------
  // Per-thread weight staging: 2 fixed chunks, pipelined global fetch.
  const int cc0 = tid, cc1 = tid + 128;
  const float* ws0 =
      Wfc0 + (size_t)(((cc0 & 31) >> 4) * 16) * HID_ + (cc0 >> 5) * 16 + (cc0 & 15);
  const float* ws1 =
      Wfc0 + (size_t)(((cc1 & 31) >> 4) * 16) * HID_ + (cc1 >> 5) * 16 + (cc1 & 15);
  bf16* wd0 = &sWfrag[cc0 * 16];
  bf16* wd1 = &sWfrag[cc1 * 16];
  // A-tile copy coordinates (bf16 mirror path, non-TDM fallback)
  const int ar = tid >> 2, ach = (tid & 3) * 8;
#if USE_TDM
  const unsigned int sA_lds = (unsigned int)(unsigned long long)&sA[0][0];
#endif

  float f0[16], f1[16];
#pragma unroll
  for (int p = 0; p < 16; ++p) {
    f0[p] = ws0[(size_t)p * HID_];
    f1[p] = ws1[(size_t)p * HID_];
  }

  v8f acc[4] = {};
  for (int k0 = 0; k0 < IN_SIZE_; k0 += 32) {
    // A tile into LDS
    if (has_mirror) {
#if USE_TDM
      if (wave == 0)
        tdm_load_a_tile(sA_lds, bch_in + (size_t)mBase * IN_SIZE_ + k0);
#else
      *(v8bf*)&sA[ar][ach] =
          *(const v8bf*)(bch_in + (size_t)(mBase + ar) * IN_SIZE_ + k0 + ach);
#endif
    } else {
      for (int t = tid; t < 32 * 16; t += 128) {
        int r = t >> 4, cp = (t & 15) * 2;
        const float* s = &bc_in[(size_t)(mBase + r) * IN_SIZE_ + k0 + cp];
        v2bf v;
        v[0] = (bf16)s[0];
        v[1] = (bf16)s[1];
        *(v2bf*)&sA[r][cp] = v;
      }
    }
    // Store current weight regs -> LDS (fragment order)
    {
      v8bf a0, b0, a1, b1;
#pragma unroll
      for (int p = 0; p < 8; ++p) {
        a0[p] = (bf16)f0[p];
        b0[p] = (bf16)f0[p + 8];
        a1[p] = (bf16)f1[p];
        b1[p] = (bf16)f1[p + 8];
      }
      *(v8bf*)wd0 = a0;
      *(v8bf*)(wd0 + 8) = b0;
      *(v8bf*)wd1 = a1;
      *(v8bf*)(wd1 + 8) = b1;
    }
    // Prefetch next K-step weights into registers (overlaps WMMA below)
    if (k0 + 32 < IN_SIZE_) {
      const float* p0 = ws0 + (size_t)(k0 + 32) * HID_;
      const float* p1 = ws1 + (size_t)(k0 + 32) * HID_;
#pragma unroll
      for (int p = 0; p < 16; ++p) {
        f0[p] = p0[(size_t)p * HID_];
        f1[p] = p1[(size_t)p * HID_];
      }
    }
#if USE_TDM
    if (has_mirror && wave == 0) __builtin_amdgcn_s_wait_tensorcnt(0);
#endif
    __syncthreads();
    v16bf a = load_a_frag(&sA[rowTile * 16 + l][0], hi);
#pragma unroll
    for (int ct = 0; ct < 4; ++ct) {
      v16bf b = load_b_frag(&sWfrag[((ct0 + ct) * 32 + lane) * 16]);
      acc[ct] = wmma_bf16(a, b, acc[ct]);
    }
    __syncthreads();
  }
#pragma unroll
  for (int ct = 0; ct < 4; ++ct) {
    int col = (ct0 + ct) * 16 + l;
    float bias = bfc0[col];
#pragma unroll
    for (int v = 0; v < 8; ++v) {
      int r = rowTile * 16 + v + 8 * hi;
      float val = acc[ct][v] + bias;
      sX[0][r][col] = (bf16)(val > 0.f ? val : 0.f);
    }
  }
  __syncthreads();

  // ---------------- Stages 2-3: two 128x128 hidden layers ------------------
  int cur = 0;
  for (int layer = 0; layer < 2; ++layer) {
    const float* W = Wfc + layer * HID_ * HID_;
    const float* bias = bfc + layer * HID_;
    for (int cc = tid; cc < 1024; cc += 128) {
      int kstep = cc >> 8, ct = (cc >> 5) & 7, ln = cc & 31;
      int n = ln & 15, kh = ln >> 4;
      fill_chunk16(&sWfrag[cc * 16],
                   W + (size_t)(kstep * 32 + kh * 16) * HID_ + ct * 16 + n,
                   HID_);
    }
    __syncthreads();
    v8f acc2[4] = {};
    for (int kstep = 0; kstep < 4; ++kstep) {
      v16bf a = load_a_frag(&sX[cur][rowTile * 16 + l][kstep * 32], hi);
#pragma unroll
      for (int ct = 0; ct < 4; ++ct) {
        v16bf b =
            load_b_frag(&sWfrag[((kstep * 8 + ct0 + ct) * 32 + lane) * 16]);
        acc2[ct] = wmma_bf16(a, b, acc2[ct]);
      }
    }
#pragma unroll
    for (int ct = 0; ct < 4; ++ct) {
      int col = (ct0 + ct) * 16 + l;
      float bv = bias[col];
#pragma unroll
      for (int v = 0; v < 8; ++v) {
        int r = rowTile * 16 + v + 8 * hi;
        float val = acc2[ct][v] + bv;
        sX[1 - cur][r][col] = (bf16)(val > 0.f ? val : 0.f);
      }
    }
    __syncthreads();
    cur ^= 1;
  }

  // ---------------- Stage 4: forecast accumulation (128 -> 12) -------------
  for (int t = tid; t < 32 * HOR_; t += 128) {
    int r = t / HOR_, c = t % HOR_;
    float s = bfv[c];
    for (int k = 0; k < HID_; k += 8) {
      v8bf xv = *(const v8bf*)&sX[cur][r][k];
#pragma unroll
      for (int i = 0; i < 8; ++i) s += (float)xv[i] * Wf[(k + i) * HOR_ + c];
    }
    facc[(size_t)(mBase + r) * HOR_ + c] += s;
  }

  // ---------------- Stage 5: backcast = relu(backcast - x@Wb - bb) ---------
  for (int c0 = 0; c0 < IN_SIZE_; c0 += 128) {
    __syncthreads();
    for (int cc = tid; cc < 1024; cc += 128) {
      int kstep = cc >> 8, ct = (cc >> 5) & 7, ln = cc & 31;
      int n = ln & 15, kh = ln >> 4;
      fill_chunk16(&sWfrag[cc * 16],
                   Wb + (size_t)(kstep * 32 + kh * 16) * IN_SIZE_ + c0 +
                       ct * 16 + n,
                   IN_SIZE_);
    }
    __syncthreads();
    v8f acc3[4] = {};
    for (int kstep = 0; kstep < 4; ++kstep) {
      v16bf a = load_a_frag(&sX[cur][rowTile * 16 + l][kstep * 32], hi);
#pragma unroll
      for (int ct = 0; ct < 4; ++ct) {
        v16bf b =
            load_b_frag(&sWfrag[((kstep * 8 + ct0 + ct) * 32 + lane) * 16]);
        acc3[ct] = wmma_bf16(a, b, acc3[ct]);
      }
    }
#pragma unroll
    for (int ct = 0; ct < 4; ++ct) {
      int gcol = c0 + (ct0 + ct) * 16 + l;
      float bias = bb[gcol];
#pragma unroll
      for (int v = 0; v < 8; ++v) {
        int grow = mBase + rowTile * 16 + v + 8 * hi;
        size_t off = (size_t)grow * IN_SIZE_ + gcol;
        float val = bc_in[off] - acc3[ct][v] - bias;
        float rv = val > 0.f ? val : 0.f;
        bc_out[off] = rv;
        if (bch_out) bch_out[off] = (bf16)rv;
      }
    }
  }
}

// ---------------------------------------------------------------------------
// Kernel 6: forecast = (1 + tgf) * facc * level
// ---------------------------------------------------------------------------
__global__ __launch_bounds__(256) void forecast_kernel(
    const float* __restrict__ tgf, const float* __restrict__ facc,
    const float* __restrict__ level, float* __restrict__ out_f) {
  int idx = blockIdx.x * blockDim.x + threadIdx.x;
  if (idx >= ROWS_ * HOR_) return;
  int row = idx / HOR_;
  out_f[idx] = (1.0f + tgf[idx]) * (facc[idx] * level[row]);
}

// ---------------------------------------------------------------------------
extern "C" void kernel_launch(void* const* d_in, const int* in_sizes, int n_in,
                              void* d_out, int out_size, void* d_ws, size_t ws_size,
                              hipStream_t stream) {
  const float* history = (const float*)d_in[0];
  const float* tod     = (const float*)d_in[1];
  const int*   nid     = (const int*)d_in[2];
  const float* emb     = (const float*)d_in[3];
  const float* Wg1     = (const float*)d_in[4];
  const float* bg1     = (const float*)d_in[5];
  const float* Wg2     = (const float*)d_in[6];
  const float* bg2     = (const float*)d_in[7];
  const float* Wg3     = (const float*)d_in[8];
  const float* bg3     = (const float*)d_in[9];
  const float* Wfc0    = (const float*)d_in[10];
  const float* bfc0    = (const float*)d_in[11];
  const float* Wfc     = (const float*)d_in[12];
  const float* bfc     = (const float*)d_in[13];
  const float* Wf      = (const float*)d_in[14];
  const float* bfv     = (const float*)d_in[15];
  const float* Wb      = (const float*)d_in[16];
  const float* bb      = (const float*)d_in[17];

  float* ws    = (float*)d_ws;
  float* tg    = ws;                          // ROWS*HID
  float* tgf   = tg   + (size_t)ROWS_ * HID_;
  float* tgb   = tgf  + (size_t)ROWS_ * HOR_;
  float* hist  = tgb  + (size_t)ROWS_ * H_;
  float* level = hist + (size_t)ROWS_ * H_;
  float* dp    = level + ROWS_;
  float* facc  = dp   + N_ * N_;
  size_t float_bytes = ((size_t)(facc + (size_t)ROWS_ * HOR_) - (size_t)ws);
  size_t mirror_off  = (float_bytes + 255) & ~(size_t)255;
  size_t mirror_bytes = (size_t)ROWS_ * IN_SIZE_ * sizeof(bf16);
  bool use_mirror = (ws_size >= mirror_off + mirror_bytes);
  bf16* bch = use_mirror ? (bf16*)((char*)d_ws + mirror_off) : nullptr;

  float* out      = (float*)d_out;
  float* backcast = out;                                   // ROWS x IN_SIZE
  float* forecast = out + (size_t)ROWS_ * IN_SIZE_;        // ROWS x HOR

  gates1_kernel<<<ROWS_, 128, 0, stream>>>(emb, nid, tod, Wg1, bg1, tg);
  gates2_kernel<<<(ROWS_ + 127) / 128, 128, 0, stream>>>(
      tg, Wg2, bg2, Wg3, bg3, history, tgf, tgb, hist, level, facc);
  dp_kernel<<<(N_ * N_ + 255) / 256, 256, 0, stream>>>(emb, nid, dp);
  hbuild_kernel<<<ROWS_, 256, 0, stream>>>(hist, level, dp, emb, nid, backcast,
                                           bch);

  for (int i = 0; i < 3; ++i) {
    bf16* bch_out = (i < 2) ? bch : nullptr;   // last block has no consumer
    block_kernel<<<ROWS_ / 32, 128, 0, stream>>>(
        backcast, backcast, bch, bch_out,
        Wfc0 + (size_t)i * IN_SIZE_ * HID_, bfc0 + i * HID_,
        Wfc  + (size_t)i * 2 * HID_ * HID_, bfc  + i * 2 * HID_,
        Wf   + (size_t)i * HID_ * HOR_,     bfv  + i * HOR_,
        Wb   + (size_t)i * HID_ * IN_SIZE_, bb   + i * IN_SIZE_,
        facc);
  }

  forecast_kernel<<<(ROWS_ * HOR_ + 255) / 256, 256, 0, stream>>>(
      tgf, facc, level, forecast);
}
